// MACE_17815524344052
// MI455X (gfx1250) — matrix operand order repaired and checked by reading, not compiled, not run on hardware
//
#include <hip/hip_runtime.h>
#include <hip/hip_bf16.h>
#include <math.h>

// ---------------------------------------------------------------------------
// MACE-style equivariant GNN for MI455X (gfx1250, wave32, WMMA).
// Dense channel mixes run on v_wmma_f32_16x16x32_bf16 (bf16 in, f32 acc).
// Weights are pre-converted to bf16 and pre-transposed to [n][k] so both
// A and B fragments are contiguous bf16 runs (b128 loads, no inner converts).
// Edge gather/scatter is the memory-bound part: thread = (edge, channel),
// channel maps to lanes -> coalesced 512B rows for gather and f32 atomics.
// ---------------------------------------------------------------------------

typedef __attribute__((ext_vector_type(16))) __bf16 v16bf;
typedef __attribute__((ext_vector_type(8)))  float  v8f;
typedef __bf16 bf16_t;

#define Nn 4000
#define Ee 48000
#define Aa 10
#define Cc 128
#define Kk 16
#define Hh 64
#define Bb 8
#define NKC (Nn * Kk * Cc)   // 8,192,000

__constant__ int c_lidx[16] = {0,1,1,1,2,2,2,2,2,3,3,3,3,3,3,3};

__device__ __forceinline__ float fast_silu(float x)
{
    // x / (1 + e^-x) with v_rcp_f32 instead of IEEE division
    return x * __builtin_amdgcn_rcpf(1.0f + __expf(-x));
}

// ---------------------------------------------------------------------------
// Edge geometry: spherical harmonics Y[e][16] and radial basis Rb[e][8]
// ---------------------------------------------------------------------------
__global__ void geom_kernel(const float* __restrict__ coords,
                            const int* __restrict__ eidx,
                            float* __restrict__ Y, float* __restrict__ Rb)
{
    int e = blockIdx.x * blockDim.x + threadIdx.x;
    if (e >= Ee) return;
    int s = eidx[e], d = eidx[Ee + e];
    float dx = coords[s*3+0] - coords[d*3+0];
    float dy = coords[s*3+1] - coords[d*3+1];
    float dz = coords[s*3+2] - coords[d*3+2];
    float r  = sqrtf(dx*dx + dy*dy + dz*dz + 1e-12f);
    float inv = 1.0f / r;
    float x = dx*inv, y = dy*inv, z = dz*inv;
    float x2 = x*x, y2 = y*y, z2 = z*z;

    const float s3 = 1.7320508075688772f, s5 = 2.23606797749979f;
    const float s15 = 3.872983346207417f, s105 = 10.246950765959598f;
    const float c3m3 = 2.091650066335189f, c3m1 = 1.620185174601965f;
    const float c30 = 1.3228756555322954f;

    float* yr = Y + (size_t)e * 16;
    yr[0]  = 1.0f;
    yr[1]  = s3 * x;   yr[2] = s3 * y;   yr[3] = s3 * z;
    yr[4]  = s15 * x * y;
    yr[5]  = s15 * y * z;
    yr[6]  = 0.5f * s5 * (3.0f * z2 - 1.0f);
    yr[7]  = s15 * x * z;
    yr[8]  = 0.5f * s15 * (x2 - y2);
    yr[9]  = c3m3 * y * (3.0f * x2 - y2);
    yr[10] = 0.5f * s105 * x * y * z;
    yr[11] = c3m1 * y * (5.0f * z2 - 1.0f);
    yr[12] = c30  * z * (5.0f * z2 - 3.0f);
    yr[13] = c3m1 * x * (5.0f * z2 - 1.0f);
    yr[14] = 0.25f * s105 * z * (x2 - y2);
    yr[15] = c3m3 * x * (x2 - 3.0f * y2);

    // radial bessel * polynomial cutoff (p = 5)
    float t = r * 0.2f;                       // r / R_MAX
    float fc = 0.0f;
    if (t < 1.0f) {
        float t5 = t*t*t*t*t;
        fc = 1.0f - 21.0f*t5 + 35.0f*t5*t - 15.0f*t5*t*t;
    }
    float amp = 0.6324555320336759f * inv * fc;   // sqrt(2/R_MAX)/r * fcut
    const float pi = 3.14159265358979323846f;
    float* rr = Rb + (size_t)e * 8;
    #pragma unroll
    for (int n = 1; n <= 8; ++n)
        rr[n-1] = amp * sinf((float)n * pi * t);
}

// ---------------------------------------------------------------------------
// Node embedding: feats[n][0][c] = attrs@W_embed, feats[n][k>0][c] = 0
// Also attr[n][c] = attrs@W_attr (used by layer-0 gating).
// ---------------------------------------------------------------------------
__global__ void embed_kernel(const float* __restrict__ attrs,
                             const float* __restrict__ W_embed,
                             const float* __restrict__ W_attr,
                             float* __restrict__ F, float* __restrict__ attrb)
{
    int tid = blockIdx.x * blockDim.x + threadIdx.x;
    if (tid >= Nn * Cc) return;
    int n = tid >> 7, c = tid & 127;
    float e = 0.0f, a = 0.0f;
    #pragma unroll
    for (int j = 0; j < Aa; ++j) {
        float na = attrs[n*Aa + j];
        e += na * W_embed[j*Cc + c];
        a += na * W_attr[j*Cc + c];
    }
    float* fr = F + (size_t)n * (Kk*Cc) + c;
    fr[0] = e;
    #pragma unroll
    for (int k = 1; k < Kk; ++k) fr[k*Cc] = 0.0f;
    attrb[tid] = a;
}

// ---------------------------------------------------------------------------
// Weight prep: transpose-convert f32 [K x N] -> bf16 [N x K]
// ---------------------------------------------------------------------------
__global__ void wcvt_kernel(const float* __restrict__ in, bf16_t* __restrict__ out,
                            int Kd, int Ncols)
{
    int tid = blockIdx.x * blockDim.x + threadIdx.x;
    if (tid >= Kd * Ncols) return;
    int n = tid / Kd, k = tid % Kd;
    out[tid] = (bf16_t)in[(size_t)k * Ncols + n];
}

// ---------------------------------------------------------------------------
// Radial MLP layer 1 (K=8, too small for WMMA): h1 = silu(R @ Wr1), bf16 out
// ---------------------------------------------------------------------------
__global__ void r1_kernel(const float* __restrict__ Rb,
                          const float* __restrict__ Wr1,
                          bf16_t* __restrict__ h1)
{
    int e = blockIdx.x * blockDim.x + threadIdx.x;
    if (e >= Ee) return;
    float rb[8];
    #pragma unroll
    for (int b = 0; b < 8; ++b) rb[b] = Rb[(size_t)e*8 + b];
    bf16_t* hr = h1 + (size_t)e * Hh;
    for (int h = 0; h < Hh; ++h) {
        float s = 0.0f;
        #pragma unroll
        for (int b = 0; b < 8; ++b) s += rb[b] * Wr1[b*Hh + h];
        hr[h] = (bf16_t)fast_silu(s);
    }
}

// ---------------------------------------------------------------------------
// Generic wave-tile GEMM:  D[M x Nc] = A(bf16)[M x Kd] * Bt(bf16)[Nc x Kd]^T
// One wave per 16x16 tile, K stepped by 32 via v_wmma_f32_16x16x32_bf16.
// k order is contiguous (k = 16*half + i) in A and B; both fragments are
// contiguous bf16 runs -> global_load_b128.
// flags: bit0 = silu, bit1 = bf16 output
// ---------------------------------------------------------------------------
__global__ void gemm_wmma_kernel(const bf16_t* __restrict__ A, int lda,
                                 const bf16_t* __restrict__ Bt,
                                 void* __restrict__ Dp, int ldd,
                                 int M, int Ncols, int Kd, int flags)
{
    const int waveId = threadIdx.x >> 5;
    const int lane   = threadIdx.x & 31;
    const int half   = lane >> 4;
    const int l      = lane & 15;
    const int nTiles = Ncols >> 4;
    const int total  = (M >> 4) * nTiles;
    int t = blockIdx.x * (blockDim.x >> 5) + waveId;
    if (t >= total) return;
    const int m0 = (t / nTiles) << 4;
    const int n0 = (t % nTiles) << 4;

    v8f acc = {};
    const bf16_t* arow = A  + (size_t)(m0 + l) * lda + half * 16;
    const bf16_t* brow = Bt + (size_t)(n0 + l) * Kd  + half * 16;
    for (int k0 = 0; k0 < Kd; k0 += 32) {
        v16bf af, bfv;
        // A 16x32 frag: lanes 0-15 hold K 0..15, lanes 16-31 K 16..31; one contiguous 16-run
        const bf16_t* ap = arow + k0;
        #pragma unroll
        for (int i = 0; i < 16; ++i) af[i] = ap[i];
        // B 32x16 frag: lanes 0-15 hold K 0..15, lanes 16-31 K 16..31;
        // Bt is [n][k] so this is one contiguous 16-element run
        const bf16_t* bp = brow + k0;
        #pragma unroll
        for (int i = 0; i < 16; ++i) bfv[i] = bp[i];
        acc = __builtin_amdgcn_wmma_f32_16x16x32_bf16(
                  false, af, false, bfv, (short)0, acc, false, false);
    }

    const int row = m0 + half * 8;
    const int col = n0 + l;
    if (flags & 2) {
        bf16_t* D = (bf16_t*)Dp;
        #pragma unroll
        for (int j = 0; j < 8; ++j) {
            float v = acc[j];
            if (flags & 1) v = fast_silu(v);
            D[(size_t)(row + j) * ldd + col] = (bf16_t)v;
        }
    } else {
        float* D = (float*)Dp;
        #pragma unroll
        for (int j = 0; j < 8; ++j) {
            float v = acc[j];
            if (flags & 1) v = fast_silu(v);
            D[(size_t)(row + j) * ldd + col] = v;
        }
    }
}

// ---------------------------------------------------------------------------
// Final fused GEMM: out = m[:, k=0, :] @ W_msg1 + feats[:, k=0, :] @ W_skip
// A slices are f32 rows with stride K*C (k=0 planes), converted in-register;
// weights come in pre-transposed bf16.
// ---------------------------------------------------------------------------
__global__ void gemm2_wmma_kernel(const float*  __restrict__ A1,
                                  const float*  __restrict__ A2,
                                  const bf16_t* __restrict__ B1t,
                                  const bf16_t* __restrict__ B2t,
                                  float* __restrict__ D)
{
    const int waveId = threadIdx.x >> 5;
    const int lane   = threadIdx.x & 31;
    const int half   = lane >> 4;
    const int l      = lane & 15;
    const int nTiles = Cc >> 4;                    // 8
    const int total  = (Nn >> 4) * nTiles;         // 2000
    int t = blockIdx.x * (blockDim.x >> 5) + waveId;
    if (t >= total) return;
    const int m0 = (t / nTiles) << 4;
    const int n0 = (t % nTiles) << 4;
    const int lda = Kk * Cc;                       // k=0 plane row stride

    v8f acc = {};
    #pragma unroll 1
    for (int pass = 0; pass < 2; ++pass) {
        const float*  A  = pass ? A2  : A1;
        const bf16_t* Bt = pass ? B2t : B1t;
        const float*  arow = A  + (size_t)(m0 + l) * lda + half * 16;
        const bf16_t* brow = Bt + (size_t)(n0 + l) * Cc  + half * 16;
        for (int k0 = 0; k0 < Cc; k0 += 32) {
            v16bf af, bfv;
            const float* ap = arow + k0;
            #pragma unroll
            for (int i = 0; i < 16; ++i) {
                af[i]   = (bf16_t)ap[i];
            }
            const bf16_t* bp = brow + k0;
            #pragma unroll
            for (int i = 0; i < 16; ++i) bfv[i] = bp[i];
            acc = __builtin_amdgcn_wmma_f32_16x16x32_bf16(
                      false, af, false, bfv, (short)0, acc, false, false);
        }
    }
    const int row = m0 + half * 8;
    const int col = n0 + l;
    #pragma unroll
    for (int j = 0; j < 8; ++j)
        D[(size_t)(row + j) * Cc + col] = acc[j];
}

// ---------------------------------------------------------------------------
// f32 -> bf16 convert, zero fill
// ---------------------------------------------------------------------------
__global__ void cvt_kernel(const float* __restrict__ in, bf16_t* __restrict__ out, int n)
{
    int i = blockIdx.x * blockDim.x + threadIdx.x;
    if (i < n) out[i] = (bf16_t)in[i];
}
__global__ void zero_kernel(float* __restrict__ p, int n)
{
    int i = blockIdx.x * blockDim.x + threadIdx.x;
    if (i < n) p[i] = 0.0f;
}

// ---------------------------------------------------------------------------
// Message + scatter: per (edge, channel):
//   g = sum_k hu[src][k][c] * Y[e][k]
//   acc[dst][k][c] += w[e][c*4 + l_idx[k]] * Y[e][k] * g / AVG_NEI
// Channel maps to lanes -> coalesced gather rows and coalesced atomics.
// ---------------------------------------------------------------------------
__global__ void msg_kernel(const float* __restrict__ hu,
                           const bf16_t* __restrict__ w,
                           const float* __restrict__ Y,
                           const int* __restrict__ eidx,
                           float* __restrict__ acc)
{
    int tid = blockIdx.x * blockDim.x + threadIdx.x;
    if (tid >= Ee * Cc) return;
    int e = tid >> 7, c = tid & 127;
    int s = eidx[e], d = eidx[Ee + e];

    const float* yr = Y + (size_t)e * 16;
    const bf16_t* wr = w + (size_t)e * (Cc*4) + c*4;
    // hint the streaming per-edge rows (global_prefetch_b8)
    __builtin_prefetch(yr + 32, 0, 0);
    __builtin_prefetch(wr + Cc*4*2, 0, 0);

    float y[16];
    #pragma unroll
    for (int k = 0; k < 16; ++k) y[k] = yr[k];

    const float* hr = hu + (size_t)s * (Kk*Cc) + c;
    float g = 0.0f;
    #pragma unroll
    for (int k = 0; k < 16; ++k) g += hr[k*Cc] * y[k];

    float w4[4];
    #pragma unroll
    for (int j = 0; j < 4; ++j) w4[j] = (float)wr[j];

    float gs = g * (1.0f / 12.0f);   // / AVG_NEI
    float* ar = acc + (size_t)d * (Kk*Cc) + c;
    #pragma unroll
    for (int k = 0; k < 16; ++k)
        atomicAdd(&ar[k*Cc], w4[c_lidx[k]] * y[k] * gs);
}

// ---------------------------------------------------------------------------
// Per-node polynomial gate (in place on a2 = [n][k][c]):
//   layer0: a *= attr;  s = a[:,k=0,:];  a *= (w0 + w1*s + w2*s^2)
// ---------------------------------------------------------------------------
__global__ void poly_kernel(float* __restrict__ a2,
                            const float* __restrict__ attrb,
                            const float* __restrict__ w_corr_i,
                            int layer0)
{
    int tid = blockIdx.x * blockDim.x + threadIdx.x;
    if (tid >= Nn * Cc) return;
    int n = tid >> 7, c = tid & 127;
    float* row = a2 + (size_t)n * (Kk*Cc) + c;
    if (layer0) {
        float av = attrb[tid];
        #pragma unroll
        for (int k = 0; k < Kk; ++k) row[k*Cc] *= av;
    }
    float s = row[0];
    float p = w_corr_i[c] + w_corr_i[Cc + c] * s + w_corr_i[2*Cc + c] * s * s;
    #pragma unroll
    for (int k = 0; k < Kk; ++k) row[k*Cc] *= p;
}

// ---------------------------------------------------------------------------
// Write outs[0]: transpose feats [n][k][c] -> d_out [n][c][k]
// ---------------------------------------------------------------------------
__global__ void writeout_kernel(const float* __restrict__ F, float* __restrict__ out)
{
    int tid = blockIdx.x * blockDim.x + threadIdx.x;
    if (tid >= NKC) return;
    int n = tid >> 11;
    int c = (tid >> 4) & 127;
    int k = tid & 15;
    out[tid] = F[(size_t)n * (Kk*Cc) + k*Cc + c];
}

// ---------------------------------------------------------------------------
// Host-side orchestration
// ---------------------------------------------------------------------------
extern "C" void kernel_launch(void* const* d_in, const int* in_sizes, int n_in,
                              void* d_out_, int out_size, void* d_ws, size_t ws_size,
                              hipStream_t stream)
{
    const float* coords  = (const float*)d_in[0];
    const float* attrs   = (const float*)d_in[1];
    const float* W_embed = (const float*)d_in[2];
    const float* Wr1     = (const float*)d_in[3];
    const float* Wr2     = (const float*)d_in[4];
    const float* Wr3     = (const float*)d_in[5];
    const float* W_up    = (const float*)d_in[6];
    const float* W_out   = (const float*)d_in[7];
    const float* W_attr  = (const float*)d_in[8];
    const float* w_corr  = (const float*)d_in[9];
    const float* W_msg   = (const float*)d_in[10];
    const float* W_skip  = (const float*)d_in[11];
    const int*   eidx    = (const int*)d_in[12];
    float* dout = (float*)d_out_;

    // workspace carve-up (~182 MB)
    char* ws = (char*)d_ws;
    size_t off = 0;
    auto take = [&](size_t bytes) -> char* {
        char* p = ws + off;
        off = (off + bytes + 255) & ~(size_t)255;
        return p;
    };
    float*  F    = (float*) take(sizeof(float)  * NKC);        // feats  [n][k][c]
    float*  T0   = (float*) take(sizeof(float)  * NKC);        // hu / a2 / m
    float*  T1   = (float*) take(sizeof(float)  * NKC);        // scatter accumulator
    float*  Ybuf = (float*) take(sizeof(float)  * Ee * 16);
    float*  Rbuf = (float*) take(sizeof(float)  * Ee * 8);
    bf16_t* h1   = (bf16_t*)take(sizeof(bf16_t) * Ee * Hh);
    bf16_t* h2   = (bf16_t*)take(sizeof(bf16_t) * Ee * Hh);
    bf16_t* wbuf = (bf16_t*)take(sizeof(bf16_t) * (size_t)Ee * Cc * 4);
    bf16_t* Abf  = (bf16_t*)take(sizeof(bf16_t) * NKC);        // bf16 GEMM staging
    float*  attrb= (float*) take(sizeof(float)  * Nn * Cc);
    // pre-transposed bf16 weights (per layer where applicable)
    bf16_t* Wr2t  = (bf16_t*)take(sizeof(bf16_t) * 2 * Hh * Hh);
    bf16_t* Wr3t  = (bf16_t*)take(sizeof(bf16_t) * 2 * Hh * Cc * 4);
    bf16_t* Wupt  = (bf16_t*)take(sizeof(bf16_t) * 2 * Cc * Cc);
    bf16_t* Woutt = (bf16_t*)take(sizeof(bf16_t) * 2 * Cc * Cc);
    bf16_t* Wmsgt = (bf16_t*)take(sizeof(bf16_t) * 2 * Cc * Cc);
    bf16_t* Wskpt = (bf16_t*)take(sizeof(bf16_t) * Cc * Cc);

    auto wcvt = [&](const float* in, bf16_t* out, int Kd, int Ncols) {
        int n = Kd * Ncols;
        wcvt_kernel<<<(n + 255)/256, 256, 0, stream>>>(in, out, Kd, Ncols);
    };
    auto gemm = [&](const bf16_t* A, int lda, const bf16_t* Bt,
                    void* D, int ldd, int M, int Ncols, int Kd, int flags) {
        int tiles  = (M / 16) * (Ncols / 16);
        int blocks = (tiles + 7) / 8;                 // 8 waves / 256-thread block
        gemm_wmma_kernel<<<blocks, 256, 0, stream>>>(A, lda, Bt, D, ldd,
                                                     M, Ncols, Kd, flags);
    };

    // --- weight prep (tiny, once) ---
    for (int i = 0; i < 2; ++i) {
        wcvt(Wr2  + (size_t)i*Hh*Hh,    Wr2t  + (size_t)i*Hh*Hh,    Hh, Hh);
        wcvt(Wr3  + (size_t)i*Hh*Cc*4,  Wr3t  + (size_t)i*Hh*Cc*4,  Hh, Cc*4);
        wcvt(W_up + (size_t)i*Cc*Cc,    Wupt  + (size_t)i*Cc*Cc,    Cc, Cc);
        wcvt(W_out+ (size_t)i*Cc*Cc,    Woutt + (size_t)i*Cc*Cc,    Cc, Cc);
        wcvt(W_msg+ (size_t)i*Cc*Cc,    Wmsgt + (size_t)i*Cc*Cc,    Cc, Cc);
    }
    wcvt(W_skip, Wskpt, Cc, Cc);

    // --- edge geometry + node embed (once) ---
    geom_kernel <<<(Ee + 255)/256, 256, 0, stream>>>(coords, eidx, Ybuf, Rbuf);
    embed_kernel<<<(Nn*Cc + 255)/256, 256, 0, stream>>>(attrs, W_embed, W_attr, F, attrb);

    for (int i = 0; i < 2; ++i) {
        const float* Wr1i = Wr1 + (size_t)i * Bb * Hh;
        const float* wci  = w_corr + (size_t)i * 3 * Cc;

        // radial MLP: h1 (VALU) -> h2 (WMMA, silu) -> w (WMMA)
        r1_kernel<<<(Ee + 255)/256, 256, 0, stream>>>(Rbuf, Wr1i, h1);
        gemm(h1, Hh, Wr2t + (size_t)i*Hh*Hh,   h2,   Hh,   Ee, Hh,   Hh, /*silu|bf16*/3);
        gemm(h2, Hh, Wr3t + (size_t)i*Hh*Cc*4, wbuf, Cc*4, Ee, Cc*4, Hh, /*bf16*/2);

        // hu = feats @ W_up   (flat 64000x128x128 bf16 WMMA)
        cvt_kernel<<<(NKC + 255)/256, 256, 0, stream>>>(F, Abf, NKC);
        gemm(Abf, Cc, Wupt + (size_t)i*Cc*Cc, T0, Cc, Nn*Kk, Cc, Cc, 0);

        // scatter messages into T1
        zero_kernel<<<(NKC + 255)/256, 256, 0, stream>>>(T1, NKC);
        msg_kernel<<<(Ee*Cc + 255)/256, 256, 0, stream>>>(T0, wbuf, Ybuf, eidx, T1);

        // a2 = a @ W_out, then polynomial gate (and attr gate on layer 0)
        cvt_kernel<<<(NKC + 255)/256, 256, 0, stream>>>(T1, Abf, NKC);
        gemm(Abf, Cc, Woutt + (size_t)i*Cc*Cc, T0, Cc, Nn*Kk, Cc, Cc, 0);
        poly_kernel<<<(Nn*Cc + 255)/256, 256, 0, stream>>>(T0, attrb, wci, i == 0 ? 1 : 0);

        if (i == 0) {
            // feats = m @ W_msg[0]
            cvt_kernel<<<(NKC + 255)/256, 256, 0, stream>>>(T0, Abf, NKC);
            gemm(Abf, Cc, Wmsgt, F, Cc, Nn*Kk, Cc, Cc, 0);
        } else {
            // out = m[:,k=0,:] @ W_msg[1] + feats[:,k=0,:] @ W_skip
            int tiles  = (Nn / 16) * (Cc / 16);
            int blocks = (tiles + 7) / 8;
            gemm2_wmma_kernel<<<blocks, 256, 0, stream>>>(T0, F,
                                                          Wmsgt + (size_t)Cc*Cc, Wskpt,
                                                          dout + NKC);
        }
    }

    // outs[0] = feats after layer 0 in (n, c, k) order
    writeout_kernel<<<(NKC + 255)/256, 256, 0, stream>>>(F, dout);
}
